// CAAM_22986664969020
// MI455X (gfx1250) — matrix operand order repaired and checked by reading, not compiled, not run on hardware
//
#include <hip/hip_runtime.h>
#include <math.h>

typedef __bf16 bf16;
typedef __attribute__((ext_vector_type(16))) __bf16 v16bf;
typedef __attribute__((ext_vector_type(8)))  float  v8f;
typedef __attribute__((ext_vector_type(4)))  int    v4i;

#define SEQ_L 2048
#define BATCH 16

// ---------------------------------------------------------------------------
// CDNA5 async global->LDS copy (ASYNCcnt path), with safe sync fallback.
// Builtin proto: (v4i AS1* src, v4i AS3* dst, imm offset, imm cpol)
// ---------------------------------------------------------------------------
#if __has_builtin(__builtin_amdgcn_global_load_async_to_lds_b128) && \
    __has_builtin(__builtin_amdgcn_s_wait_asynccnt)
#define HAVE_ASYNC_LDS 1
#else
#define HAVE_ASYNC_LDS 0
#endif

__device__ __forceinline__ void cp_async_b128(void* lds_dst, const void* glb_src) {
#if HAVE_ASYNC_LDS
  __builtin_amdgcn_global_load_async_to_lds_b128(
      (__attribute__((address_space(1))) v4i*)glb_src,
      (__attribute__((address_space(3))) v4i*)lds_dst, 0, 0);
#else
  *(uint4*)lds_dst = *(const uint4*)glb_src;
#endif
}

__device__ __forceinline__ void async_wait0() {
#if HAVE_ASYNC_LDS
  __builtin_amdgcn_s_wait_asynccnt(0);
#endif
}

// ---------------------------------------------------------------------------
// WMMA helpers (CDNA5 wave32, V_WMMA_F32_16X16X32_BF16)
// ---------------------------------------------------------------------------
__device__ __forceinline__ v8f wmma_bf16(v16bf a, v16bf b, v8f c) {
  return __builtin_amdgcn_wmma_f32_16x16x32_bf16(false, a, false, b, (short)0, c,
                                                 false, false);
}

// A fragment: 16 rows (m0..) x 32 K (k0..), src row-major, leading dim ld.
// Lanes 0-15 -> K {0..7,16..23}, lanes 16-31 -> K {8..15,24..31}.
__device__ __forceinline__ v16bf load_afrag(const bf16* src, int ld, int m0, int k0,
                                            int lane) {
  const int half = lane >> 4;
  const int row  = m0 + (lane & 15);
  const bf16* p = src + (size_t)row * ld + k0 + half * 8;
  v16bf a;
#pragma unroll
  for (int j = 0; j < 4; ++j) {
    a[2 * j]         = p[2 * j];
    a[2 * j + 1]     = p[2 * j + 1];
    a[8 + 2 * j]     = p[16 + 2 * j];
    a[8 + 2 * j + 1] = p[16 + 2 * j + 1];
  }
  return a;
}

// B fragment: 32 K x 16 cols; src holds B^T row-major (src[col][k]), leading dim ld.
// Lanes 0-15 -> K {0..7,16..23}, lanes 16-31 -> K {8..15,24..31}.
__device__ __forceinline__ v16bf load_bfrag(const bf16* src, int ld, int n0, int k0,
                                            int lane) {
  const int off = (lane >> 4) * 8;
  const int col = n0 + (lane & 15);
  const bf16* p = src + (size_t)col * ld + k0 + off;
  v16bf b;
#pragma unroll
  for (int j = 0; j < 8; ++j) {
    b[j]     = p[j];
    b[8 + j] = p[16 + j];
  }
  return b;
}

__device__ __forceinline__ float gelu_tanh(float x) {
  float u = 0.7978845608028654f * (x + 0.044715f * x * x * x);
  float e = __expf(2.0f * u);
  return 0.5f * x * (1.0f + (e - 1.0f) / (e + 1.0f));
}

// ---------------------------------------------------------------------------
// Tiled GEMM: Out[M,N] = act(A[M,K] @ W[K,N] + bias),  W as bf16 Bt[N][K].
// Block 256 threads = 8 waves; tile 128x64; wave -> 32x32 (4 WMMA accums).
// Double-buffered LDS, async global->LDS prefetch of tile k+1 over compute k.
// AK: 0=A fp32, 1=A bf16.  ACT: 0=none, 1=gelu.  OK: 0=store bf16, 1=store fp32.
// ---------------------------------------------------------------------------
#define GA_LD 40  // 80B rows: 16B-aligned for async b128, bank-conflict-free reads

template <int AK, int ACT, int OK>
__global__ __launch_bounds__(256) void gemm_bias_kernel(
    const void* __restrict__ Ain, int lda, const bf16* __restrict__ Bt, int K,
    const float* __restrict__ bias, void* __restrict__ Out, int ldo, int ocol) {
  __shared__ bf16 As[2][128 * GA_LD];
  __shared__ bf16 Bs[2][64 * GA_LD];

  const int tid  = threadIdx.x;
  const int lane = tid & 31;
  const int w    = tid >> 5;
  const int wm = w >> 1, wn = w & 1;     // 4 x 2 wave grid
  const int m0 = blockIdx.x * 128;
  const int n0 = blockIdx.y * 64;

  // staging assignments
  const int arow = tid >> 1, acol = (tid & 1) * 16;   // A: 128 rows x 32 cols
  const int brow = tid >> 2, bcol = (tid & 3) * 8;    // B: 64 rows x 32 cols

  v8f acc[2][2];
  v8f vz = {};
#pragma unroll
  for (int i = 0; i < 2; ++i)
#pragma unroll
    for (int j = 0; j < 2; ++j) acc[i][j] = vz;

  auto stage = [&](int k0, int buf) {
    if (AK == 0) {
      const float* A = (const float*)Ain;
      const float* src = A + (size_t)(m0 + arow) * lda + k0 + acol;
      float4 f0 = ((const float4*)src)[0];
      float4 f1 = ((const float4*)src)[1];
      float4 f2 = ((const float4*)src)[2];
      float4 f3 = ((const float4*)src)[3];
      bf16* d = &As[buf][arow * GA_LD + acol];
      d[0]  = (bf16)f0.x; d[1]  = (bf16)f0.y; d[2]  = (bf16)f0.z; d[3]  = (bf16)f0.w;
      d[4]  = (bf16)f1.x; d[5]  = (bf16)f1.y; d[6]  = (bf16)f1.z; d[7]  = (bf16)f1.w;
      d[8]  = (bf16)f2.x; d[9]  = (bf16)f2.y; d[10] = (bf16)f2.z; d[11] = (bf16)f2.w;
      d[12] = (bf16)f3.x; d[13] = (bf16)f3.y; d[14] = (bf16)f3.z; d[15] = (bf16)f3.w;
    } else {
      const bf16* src = (const bf16*)Ain + (size_t)(m0 + arow) * lda + k0 + acol;
      cp_async_b128(&As[buf][arow * GA_LD + acol], src);
      cp_async_b128(&As[buf][arow * GA_LD + acol + 8], src + 8);
    }
    const bf16* bsrc = Bt + (size_t)(n0 + brow) * K + k0 + bcol;
    cp_async_b128(&Bs[buf][brow * GA_LD + bcol], bsrc);
  };

  const int nk = K >> 5;
  stage(0, 0);
  for (int kk = 0; kk < nk; ++kk) {
    const int cur = kk & 1;
    async_wait0();
    __syncthreads();
    if (kk + 1 < nk) stage((kk + 1) * 32, cur ^ 1);

    const bf16* as = As[cur];
    const bf16* bs = Bs[cur];
    v16bf a0 = load_afrag(as, GA_LD, wm * 32, 0, lane);
    v16bf a1 = load_afrag(as, GA_LD, wm * 32 + 16, 0, lane);
    v16bf b0 = load_bfrag(bs, GA_LD, wn * 32, 0, lane);
    v16bf b1 = load_bfrag(bs, GA_LD, wn * 32 + 16, 0, lane);
    acc[0][0] = wmma_bf16(a0, b0, acc[0][0]);
    acc[0][1] = wmma_bf16(a0, b1, acc[0][1]);
    acc[1][0] = wmma_bf16(a1, b0, acc[1][0]);
    acc[1][1] = wmma_bf16(a1, b1, acc[1][1]);
  }

  const int half = lane >> 4, l16 = lane & 15;
  const int col0 = n0 + wn * 32 + l16;
  const float bb0 = bias[col0], bb1 = bias[col0 + 16];
#pragma unroll
  for (int sub = 0; sub < 2; ++sub) {
    const int orow = m0 + wm * 32 + sub * 16 + half * 8;
#pragma unroll
    for (int r = 0; r < 8; ++r) {
      float v0 = acc[sub][0][r] + bb0;
      float v1 = acc[sub][1][r] + bb1;
      if (ACT == 1) { v0 = gelu_tanh(v0); v1 = gelu_tanh(v1); }
      size_t base = (size_t)(orow + r) * ldo + ocol;
      if (OK == 0) {
        ((bf16*)Out)[base + col0]      = (bf16)v0;
        ((bf16*)Out)[base + col0 + 16] = (bf16)v1;
      } else {
        ((float*)Out)[base + col0]      = v0;
        ((float*)Out)[base + col0 + 16] = v1;
      }
    }
  }
}

// ---------------------------------------------------------------------------
// Flash attention: softmax(Q K^T / 16) V, E=256, streaming 32-key tiles.
// Block 128 threads = 4 waves; wave owns 16 query rows x full E (16 accums).
// K tile staged via async global->LDS (overlaps the manual V transpose).
// ---------------------------------------------------------------------------
#define KS_LD 264  // 528B rows: 16B-aligned for async b128, conflict-free reads

__global__ __launch_bounds__(128) void attn_kernel(
    const bf16* __restrict__ Q, const bf16* __restrict__ Kmat,
    const bf16* __restrict__ Vmat, bf16* __restrict__ H, int ldh, int hoff) {
  __shared__ bf16 Ksh[32 * KS_LD];   // K tile row-major == B^T layout for S
  __shared__ bf16 Vt[256 * 34];      // V tile transposed: [e][k]
  __shared__ bf16 Pw[4][16 * 34];    // per-wave P staging (C-layout -> A-layout)

  const int tid = threadIdx.x, lane = tid & 31, w = tid >> 5;
  const int half = lane >> 4, l16 = lane & 15;
  const int b  = blockIdx.y;
  const int q0 = blockIdx.x * 64;
  const size_t qrow0 = (size_t)b * SEQ_L + q0 + w * 16;
  const bf16* qbase  = Q + qrow0 * 256;

  v8f o[16];
  v8f vz = {};
#pragma unroll
  for (int i = 0; i < 16; ++i) o[i] = vz;
  float mrow[8], lrow[8];
#pragma unroll
  for (int r = 0; r < 8; ++r) { mrow[r] = -3.0e38f; lrow[r] = 0.0f; }

  const int crow = tid >> 2;           // 0..31 (key row)
  const int ccol = (tid & 3) * 64;     // 64-wide column chunk

  for (int kt = 0; kt < SEQ_L; kt += 32) {
    // ---- async-stage K tile (row-major), sync-stage V tile transposed ----
    const bf16* ksrc = Kmat + ((size_t)b * SEQ_L + kt + crow) * 256 + ccol;
    const bf16* vsrc = Vmat + ((size_t)b * SEQ_L + kt + crow) * 256 + ccol;
#pragma unroll
    for (int i = 0; i < 64; i += 8)
      cp_async_b128(&Ksh[crow * KS_LD + ccol + i], ksrc + i);
#pragma unroll
    for (int i = 0; i < 64; i += 8) {
      uint4 vv = *(const uint4*)(vsrc + i);
      const bf16* pv = (const bf16*)&vv;
#pragma unroll
      for (int j = 0; j < 8; ++j) Vt[(ccol + i + j) * 34 + crow] = pv[j];
    }
    async_wait0();
    __syncthreads();

    // ---- S = Q K^T : 16 queries x 32 keys, K-dim = 256 ----
    v8f s0 = {}, s1 = {};
#pragma unroll
    for (int es = 0; es < 8; ++es) {
      v16bf aq = load_afrag(qbase, 256, 0, es * 32, lane);
      v16bf k0 = load_bfrag(Ksh, KS_LD, 0,  es * 32, lane);
      v16bf k1 = load_bfrag(Ksh, KS_LD, 16, es * 32, lane);
      s0 = wmma_bf16(aq, k0, s0);
      s1 = wmma_bf16(aq, k1, s1);
    }

    // ---- online softmax (row reductions across the 16-lane half) ----
#pragma unroll
    for (int r = 0; r < 8; ++r) {
      float x0 = s0[r] * 0.0625f;   // 1/sqrt(256)
      float x1 = s1[r] * 0.0625f;
      float mx = fmaxf(x0, x1);
#pragma unroll
      for (int d = 8; d >= 1; d >>= 1) mx = fmaxf(mx, __shfl_xor(mx, d, 32));
      float mnew  = fmaxf(mrow[r], mx);
      float alpha = __expf(mrow[r] - mnew);
      float p0 = __expf(x0 - mnew);
      float p1 = __expf(x1 - mnew);
      float rs = p0 + p1;
#pragma unroll
      for (int d = 8; d >= 1; d >>= 1) rs += __shfl_xor(rs, d, 32);
      lrow[r] = lrow[r] * alpha + rs;
      mrow[r] = mnew;
#pragma unroll
      for (int et = 0; et < 16; ++et) o[et][r] *= alpha;
      Pw[w][(r + 8 * half) * 34 + l16]      = (bf16)p0;
      Pw[w][(r + 8 * half) * 34 + 16 + l16] = (bf16)p1;
    }

    // ---- O += P V ----
    v16bf ap = load_afrag(Pw[w], 34, 0, 0, lane);
#pragma unroll
    for (int et = 0; et < 16; ++et) {
      v16bf bv = load_bfrag(Vt, 34, et * 16, 0, lane);
      o[et] = wmma_bf16(ap, bv, o[et]);
    }
    __syncthreads();
  }

  // ---- normalize and write into concat slot of H ----
#pragma unroll
  for (int r = 0; r < 8; ++r) {
    float inv = 1.0f / lrow[r];
    size_t row = (qrow0 + half * 8 + r) * (size_t)ldh + hoff;
#pragma unroll
    for (int et = 0; et < 16; ++et)
      H[row + et * 16 + l16] = (bf16)(o[et][r] * inv);
  }
}

// ---------------------------------------------------------------------------
// Weight transpose+convert: W[K,N] fp32 -> Wt[N,K] bf16
// ---------------------------------------------------------------------------
__global__ __launch_bounds__(256) void transpose_bf16_kernel(
    const float* __restrict__ W, bf16* __restrict__ Wt, int K, int N) {
  int idx = blockIdx.x * blockDim.x + threadIdx.x;
  if (idx >= K * N) return;
  int k = idx / N, n = idx % N;
  Wt[(size_t)n * K + k] = (bf16)W[(size_t)k * N + n];
}

// ---------------------------------------------------------------------------
// Classifier head: pred[M,5] = E[M,256] @ Wc[256,5] + bc   (tiny, plain VALU)
// ---------------------------------------------------------------------------
__global__ __launch_bounds__(256) void cls_kernel(const float* __restrict__ E,
                                                  const float* __restrict__ Wc,
                                                  const float* __restrict__ bc,
                                                  float* __restrict__ out, int M) {
  int i = blockIdx.x * blockDim.x + threadIdx.x;
  if (i >= M) return;
  const float* row = E + (size_t)i * 256;
  float acc[5];
#pragma unroll
  for (int c = 0; c < 5; ++c) acc[c] = bc[c];
  for (int k = 0; k < 256; ++k) {
    float x = row[k];
#pragma unroll
    for (int c = 0; c < 5; ++c) acc[c] = fmaf(x, Wc[k * 5 + c], acc[c]);
  }
#pragma unroll
  for (int c = 0; c < 5; ++c) out[(size_t)i * 5 + c] = acc[c];
}

// ---------------------------------------------------------------------------
extern "C" void kernel_launch(void* const* d_in, const int* in_sizes, int n_in,
                              void* d_out, int out_size, void* d_ws, size_t ws_size,
                              hipStream_t stream) {
  const float* score = (const float*)d_in[0];
  const float* audio = (const float*)d_in[1];
  const float* W_Qs = (const float*)d_in[2];   const float* b_Qs = (const float*)d_in[3];
  const float* W_Ka = (const float*)d_in[4];   const float* b_Ka = (const float*)d_in[5];
  const float* W_Va = (const float*)d_in[6];   const float* b_Va = (const float*)d_in[7];
  const float* W_Qa = (const float*)d_in[8];   const float* b_Qa = (const float*)d_in[9];
  const float* W_Ks = (const float*)d_in[10];  const float* b_Ks = (const float*)d_in[11];
  const float* W_Vs = (const float*)d_in[12];  const float* b_Vs = (const float*)d_in[13];
  const float* W_f1 = (const float*)d_in[14];  const float* b_f1 = (const float*)d_in[15];
  const float* W_f2 = (const float*)d_in[16];  const float* b_f2 = (const float*)d_in[17];
  const float* W_cls = (const float*)d_in[18]; const float* b_cls = (const float*)d_in[19];

  const int M = BATCH * SEQ_L;  // 32768

  char* ws = (char*)d_ws;
  size_t off = 0;
  auto alloc = [&](size_t bytes) -> void* {
    void* p = ws + off;
    off += (bytes + 255) & ~(size_t)255;
    return p;
  };

  bf16* bufQ = (bf16*)alloc((size_t)M * 256 * sizeof(bf16));
  bf16* bufK = (bf16*)alloc((size_t)M * 256 * sizeof(bf16));
  bf16* bufV = (bf16*)alloc((size_t)M * 256 * sizeof(bf16));
  bf16* Hbuf = (bf16*)alloc((size_t)M * 512 * sizeof(bf16));
  bf16* WtQs = (bf16*)alloc((size_t)512 * 256 * sizeof(bf16));
  bf16* WtKa = (bf16*)alloc((size_t)768 * 256 * sizeof(bf16));
  bf16* WtVa = (bf16*)alloc((size_t)768 * 256 * sizeof(bf16));
  bf16* WtQa = (bf16*)alloc((size_t)768 * 256 * sizeof(bf16));
  bf16* WtKs = (bf16*)alloc((size_t)512 * 256 * sizeof(bf16));
  bf16* WtVs = (bf16*)alloc((size_t)512 * 256 * sizeof(bf16));
  bf16* Wtf1 = (bf16*)alloc((size_t)512 * 256 * sizeof(bf16));
  bf16* Wtf2 = (bf16*)alloc((size_t)256 * 256 * sizeof(bf16));

  auto tpose = [&](const float* W, bf16* Wt, int K, int N) {
    int n = K * N;
    transpose_bf16_kernel<<<(n + 255) / 256, 256, 0, stream>>>(W, Wt, K, N);
  };
  tpose(W_Qs, WtQs, 512, 256);
  tpose(W_Ka, WtKa, 768, 256);
  tpose(W_Va, WtVa, 768, 256);
  tpose(W_Qa, WtQa, 768, 256);
  tpose(W_Ks, WtKs, 512, 256);
  tpose(W_Vs, WtVs, 512, 256);
  tpose(W_f1, Wtf1, 512, 256);
  tpose(W_f2, Wtf2, 256, 256);

  dim3 gg(M / 128, 256 / 64);  // (256, 4)
  dim3 ga(SEQ_L / 64, BATCH);  // (32, 16)

  // --- attention 1: score queries over audio ---
  gemm_bias_kernel<0, 0, 0><<<gg, 256, 0, stream>>>(score, 512, WtQs, 512, b_Qs, bufQ, 256, 0);
  gemm_bias_kernel<0, 0, 0><<<gg, 256, 0, stream>>>(audio, 768, WtKa, 768, b_Ka, bufK, 256, 0);
  gemm_bias_kernel<0, 0, 0><<<gg, 256, 0, stream>>>(audio, 768, WtVa, 768, b_Va, bufV, 256, 0);
  attn_kernel<<<ga, 128, 0, stream>>>(bufQ, bufK, bufV, Hbuf, 512, 0);

  // --- attention 2: audio queries over score (reuse Q/K/V buffers) ---
  gemm_bias_kernel<0, 0, 0><<<gg, 256, 0, stream>>>(audio, 768, WtQa, 768, b_Qa, bufQ, 256, 0);
  gemm_bias_kernel<0, 0, 0><<<gg, 256, 0, stream>>>(score, 512, WtKs, 512, b_Ks, bufK, 256, 0);
  gemm_bias_kernel<0, 0, 0><<<gg, 256, 0, stream>>>(score, 512, WtVs, 512, b_Vs, bufV, 256, 0);
  attn_kernel<<<ga, 128, 0, stream>>>(bufQ, bufK, bufV, Hbuf, 512, 256);

  // --- FFN ---
  bf16* hbuf = bufQ;  // bufQ dead after attention 2
  gemm_bias_kernel<1, 1, 0><<<gg, 256, 0, stream>>>(Hbuf, 512, Wtf1, 512, b_f1, hbuf, 256, 0);
  float* Eout = (float*)d_out;
  gemm_bias_kernel<1, 0, 1><<<gg, 256, 0, stream>>>(hbuf, 256, Wtf2, 256, b_f2, Eout, 256, 0);

  // --- classifier head ---
  float* pred = Eout + (size_t)M * 256;
  cls_kernel<<<(M + 255) / 256, 256, 0, stream>>>(Eout, W_cls, b_cls, pred, M);
}
